// MoEGate_44487271252359
// MI455X (gfx1250) — matrix operand order repaired and checked by reading, not compiled, not run on hardware
//
#include <hip/hip_runtime.h>
#include <hip/hip_bf16.h>
#include <stdint.h>

// ---------------------------------------------------------------------------
// MoE gating (DeepSeek-style) fused kernel for gfx1250 (MI455X).
//   logits = x @ w^T  (bf16 WMMA, fp32 accum)  -> sigmoid -> grouped top-k
// T = 32768 tokens, H = 2048, E = 256 experts, 8 groups of 32,
// top-2-per-group -> top-4 groups -> top-8 experts, normalized * 2.5.
//
// Roofline: x is 268 MB fp32, read once -> 11.6 us floor at 23.3 TB/s HBM.
// bf16 WMMA keeps the 34.4 GFLOP GEMM under the memory floor; x is loaded
// non-temporally so the 2 MB w stays resident in the 192 MB L2.
// ---------------------------------------------------------------------------

typedef __attribute__((ext_vector_type(16))) __bf16 v16bf;
typedef __attribute__((ext_vector_type(2)))  __bf16 v2bf;
typedef __attribute__((ext_vector_type(8)))  float  v8f;
typedef __attribute__((ext_vector_type(4)))  float  vf4;

#define H        2048
#define NEXP     256
#define NGROUP   8
#define GSIZE    32          // experts per group
#define TOPKG    4
#define TOPK     8
#define SCALING  2.5f

#define MTILE    128         // tokens per block
#define KSTEP    32          // bf16 WMMA K
#define KITERS   (H / KSTEP) // 64
#define NTHREADS 512         // 16 wave32 waves

// LDS row strides (bytes), padded to break bank conflicts on b128 reads
#define XS_STRIDE 80         // 32 bf16 = 64B payload + 16B pad
#define WS_STRIDE 80
#define SC_STRIDE 260        // floats per score row (256 + 4 pad)

__device__ __forceinline__ unsigned pack_bf16(float a, float b) {
#if __has_builtin(__builtin_amdgcn_cvt_pk_bf16_f32)
    union { v2bf v; unsigned u; } r;
    r.v = __builtin_amdgcn_cvt_pk_bf16_f32(a, b);
    return r.u;
#else
    // Builtin not declared on this toolchain; the hardware instruction exists
    // (CDNA5 is bf16-centric), so reach it with inline asm: one VALU op packs
    // two fp32 -> two bf16 (RNE).
    unsigned out;
    asm("v_cvt_pk_bf16_f32 %0, %1, %2" : "=v"(out) : "v"(a), "v"(b));
    return out;
    // Manual RNE fallback if the mnemonic is ever rejected:
    //   union { float f; unsigned u; } ua{a}, ub{b};
    //   unsigned xx = ua.u, yy = ub.u;
    //   xx = (xx + 0x7FFFu + ((xx >> 16) & 1u)) >> 16;
    //   yy = (yy + 0x7FFFu + ((yy >> 16) & 1u)) >> 16;
    //   return xx | (yy << 16);
#endif
}

union Frag16 { uint4 q[2]; v16bf v; };

__global__ __launch_bounds__(NTHREADS, 1)
void moe_gate_gfx1250(const float* __restrict__ x,       // [T, H]
                      const float* __restrict__ w,       // [E, H]
                      const float* __restrict__ bias_g,  // [E]
                      int*   __restrict__ out_idx,       // [T, 8]
                      float* __restrict__ out_wgt)       // [T, 8]
{
    // ---- static LDS (~161 KB of the WGP's 320 KB) ----
    __shared__ __align__(16) unsigned char xs_raw[MTILE * XS_STRIDE];   // bf16 x-tile
    __shared__ __align__(16) unsigned char ws_raw[NEXP  * WS_STRIDE];   // bf16 w-tile
    __shared__ float bias_sh[NEXP];
    __shared__ float sc[MTILE * SC_STRIDE];                             // sigmoid scores

    const int tid  = threadIdx.x;
    const int tok0 = blockIdx.x * MTILE;

    const int lane    = tid & 31;
    const int wv      = tid >> 5;      // 0..15
    const int r       = lane & 15;     // row/col within 16
    const int h       = lane >> 4;     // lane half
    const int mstripe = wv >> 1;       // 0..7 : 16-row stripe
    const int nhalf   = wv & 1;        // 0..1 : 128-expert half

    // per-thread staging coordinates (8 float4 slots per 32-float row)
    const int srow = tid >> 3;         // 0..63
    const int sc4  = tid & 7;          // 0..7

    if (tid < NEXP) bias_sh[tid] = bias_g[tid];

    const v8f vzero = {0.f, 0.f, 0.f, 0.f, 0.f, 0.f, 0.f, 0.f};
    v8f acc[8];
#pragma unroll
    for (int i = 0; i < 8; ++i) acc[i] = vzero;

    // ---- register double-buffer: x rows are streamed non-temporally so
    //      the 2 MB w matrix stays hot in L2 across all 256 blocks ----
    vf4 rx[2];   // x tile: 1024 float4 / 512 thr = 2 each
    vf4 rw[4];   // w tile: 2048 float4 / 512 thr = 4 each

#define LOAD_TILES(k0)                                                        \
    do {                                                                      \
        _Pragma("unroll")                                                     \
        for (int i = 0; i < 2; ++i) {                                         \
            const int row = srow + 64 * i;                                    \
            rx[i] = __builtin_nontemporal_load(reinterpret_cast<const vf4*>(  \
                x + (size_t)(tok0 + row) * H + (k0) + 4 * sc4));              \
        }                                                                     \
        _Pragma("unroll")                                                     \
        for (int i = 0; i < 4; ++i) {                                         \
            const int row = srow + 64 * i;                                    \
            rw[i] = *reinterpret_cast<const vf4*>(                            \
                w + (size_t)row * H + (k0) + 4 * sc4);                        \
        }                                                                     \
    } while (0)

    LOAD_TILES(0);

    for (int kt = 0; kt < KITERS; ++kt) {
        const int k0 = kt * KSTEP;

        // ---- convert staged registers to bf16 and store to LDS ----
#pragma unroll
        for (int i = 0; i < 2; ++i) {
            const int row = srow + 64 * i;
            uint2 p;
            p.x = pack_bf16(rx[i][0], rx[i][1]);
            p.y = pack_bf16(rx[i][2], rx[i][3]);
            *reinterpret_cast<uint2*>(xs_raw + row * XS_STRIDE + sc4 * 8) = p;
        }
#pragma unroll
        for (int i = 0; i < 4; ++i) {
            const int row = srow + 64 * i;
            uint2 p;
            p.x = pack_bf16(rw[i][0], rw[i][1]);
            p.y = pack_bf16(rw[i][2], rw[i][3]);
            *reinterpret_cast<uint2*>(ws_raw + row * WS_STRIDE + sc4 * 8) = p;
        }

        __syncthreads();

        // ---- issue NEXT k-tile's global loads before compute (latency
        //      hidden behind the WMMA block), plus a 2-ahead prefetch ----
        if (kt + 1 < KITERS) {
            LOAD_TILES(k0 + KSTEP);
        }
        if (kt + 2 < KITERS) {
            __builtin_prefetch(x + (size_t)(tok0 + srow) * H + k0 + 2 * KSTEP, 0, 0);
        }

        // ---- A fragment: rows [16*mstripe, +16), ISA 16-bit A layout ----
        // lane half 0: v0..3 = K0..7, v4..7 = K16..23 ; half 1: +8
        Frag16 A;
        {
            const unsigned char* xrow = xs_raw + (mstripe * 16 + r) * XS_STRIDE;
            A.q[0] = *reinterpret_cast<const uint4*>(xrow + 16 * h);
            A.q[1] = *reinterpret_cast<const uint4*>(xrow + 32 + 16 * h);
        }

        // ---- 8 expert tiles of 16 for this wave's half ----
#pragma unroll
        for (int nt = 0; nt < 8; ++nt) {
            const int e = nhalf * 128 + nt * 16 + r;   // B lane column = expert
            const unsigned char* wrow = ws_raw + e * WS_STRIDE;
            Frag16 B;
            // lane half h: K 8h..8h+7 then K 16+8h..16+8h+7 (matches A)
            B.q[0] = *reinterpret_cast<const uint4*>(wrow + 16 * h);
            B.q[1] = *reinterpret_cast<const uint4*>(wrow + 32 + 16 * h);
            acc[nt] = __builtin_amdgcn_wmma_f32_16x16x32_bf16(
                false, A.v, false, B.v, (short)0, acc[nt], false, false);
        }

        __syncthreads();
    }

    // ---- sigmoid + scatter scores to LDS (C layout: N=r, M=v+8h) ----
#pragma unroll
    for (int nt = 0; nt < 8; ++nt) {
#pragma unroll
        for (int v = 0; v < 8; ++v) {
            const float s = 1.0f / (1.0f + __expf(-acc[nt][v]));
            const int m = mstripe * 16 + v + 8 * h;
            const int e = nhalf * 128 + nt * 16 + r;
            sc[m * SC_STRIDE + e] = s;
        }
    }
    __syncthreads();

    // ---- per-token grouped top-k selection (1 thread / token) ----
    if (tid < MTILE) {
        const float* row = sc + tid * SC_STRIDE;
        const float NEG = -3.402823466e+38f;

        // group scores = sum of top-2 (score + bias) per group of 32
        float gs[NGROUP];
#pragma unroll
        for (int g = 0; g < NGROUP; ++g) {
            float t1 = NEG, t2 = NEG;
            const int base = g * GSIZE;
            for (int j = 0; j < GSIZE; ++j) {
                const float v = row[base + j] + bias_sh[base + j];
                if (v > t1)      { t2 = t1; t1 = v; }
                else if (v > t2) { t2 = v; }
            }
            gs[g] = t1 + t2;
        }

        // top-4 groups (strict > keeps lowest index on ties, like lax.top_k)
        unsigned gsel = 0u;
#pragma unroll
        for (int it = 0; it < TOPKG; ++it) {
            int bg = 0; float bv = NEG;
#pragma unroll
            for (int g = 0; g < NGROUP; ++g) {
                if (!((gsel >> g) & 1u) && gs[g] > bv) { bv = gs[g]; bg = g; }
            }
            gsel |= 1u << bg;
        }

        // top-8 experts over selected groups (ascending e scan, strict >)
        float tv[TOPK]; int ti[TOPK];
#pragma unroll
        for (int i = 0; i < TOPK; ++i) { tv[i] = NEG; ti[i] = 0; }
        for (int g = 0; g < NGROUP; ++g) {
            if (!((gsel >> g) & 1u)) continue;
            const int base = g * GSIZE;
            for (int j = 0; j < GSIZE; ++j) {
                const int e   = base + j;
                const float v = row[e] + bias_sh[e];
                if (v > tv[TOPK - 1]) {
                    int p = TOPK - 1;
                    while (p > 0 && v > tv[p - 1]) {
                        tv[p] = tv[p - 1]; ti[p] = ti[p - 1]; --p;
                    }
                    tv[p] = v; ti[p] = e;
                }
            }
        }

        // weights from raw sigmoid scores, normalized, scaled
        float wgt[TOPK]; float sum = 0.0f;
#pragma unroll
        for (int i = 0; i < TOPK; ++i) { wgt[i] = row[ti[i]]; sum += wgt[i]; }
        const float scale = SCALING / (sum + 1e-20f);

        const size_t t = (size_t)tok0 + tid;
#pragma unroll
        for (int i = 0; i < TOPK; ++i) {
            out_idx[t * TOPK + i] = ti[i];
            out_wgt[t * TOPK + i] = wgt[i] * scale;
        }
    }
}

extern "C" void kernel_launch(void* const* d_in, const int* in_sizes, int n_in,
                              void* d_out, int out_size, void* d_ws, size_t ws_size,
                              hipStream_t stream) {
    (void)n_in; (void)out_size; (void)d_ws; (void)ws_size;
    const float* x    = (const float*)d_in[0];   // [8,4096,2048] fp32
    const float* w    = (const float*)d_in[1];   // [256,2048]    fp32
    const float* bias = (const float*)d_in[2];   // [256]         fp32

    const int T = in_sizes[0] / H;               // 32768 tokens

    int*   out_idx = (int*)d_out;                        // first T*8 int32
    float* out_wgt = (float*)d_out + (size_t)T * TOPK;   // then  T*8 fp32

    dim3 grid(T / MTILE);
    dim3 block(NTHREADS);
    moe_gate_gfx1250<<<grid, block, 0, stream>>>(x, w, bias, out_idx, out_wgt);
}
